// RNNCellModel_44873818309131
// MI455X (gfx1250) — matrix operand order repaired and checked by reading, not compiled, not run on hardware
//
#include <hip/hip_runtime.h>
#include <hip/hip_bf16.h>
#include <math.h>

#define B_  256
#define S_  512
#define P_  128
#define I_  128
#define H_  1024
#define KTOT (I_ + H_)      // 1152
#define NKC  (KTOT / 32)    // 36 K-chunks of 32

typedef __attribute__((ext_vector_type(16))) __bf16         bf16x16;
typedef __attribute__((ext_vector_type(8)))  float          floatx8;
typedef __attribute__((ext_vector_type(8)))  unsigned short ushortx8;
typedef __attribute__((ext_vector_type(4)))  int            v4i;

union FragU { ushortx8 h[2]; bf16x16 v; };

#define GAS __attribute__((address_space(1)))
#define LAS __attribute__((address_space(3)))

#if defined(__has_builtin)
#  if __has_builtin(__builtin_amdgcn_global_load_async_to_lds_b128)
#    define HAVE_ASYNC_LDS 1
#  endif
#  if __has_builtin(__builtin_amdgcn_s_wait_asynccnt)
#    define HAVE_WAIT_ASYNC 1
#  endif
#endif

__device__ __forceinline__ void wait_async_lds() {
#if defined(HAVE_ASYNC_LDS)
#  if defined(HAVE_WAIT_ASYNC)
    __builtin_amdgcn_s_wait_asynccnt(0);
#  else
    asm volatile("s_wait_asynccnt 0" ::: "memory");
#  endif
#endif
}

// Copy 16 bytes global(bf16) -> LDS. gfx1250 async-to-LDS path (ASYNCcnt),
// falls back to a vectorized register round-trip if the builtin is absent.
__device__ __forceinline__ void cp16_bf(const unsigned short* __restrict__ g,
                                        unsigned short* l) {
#if defined(HAVE_ASYNC_LDS)
    __builtin_amdgcn_global_load_async_to_lds_b128((GAS v4i*)g, (LAS v4i*)l, 0, 0);
#else
    *(ushortx8*)l = *(const ushortx8*)g;
#endif
}

__device__ __forceinline__ unsigned short f2bf(float f) {
    union { float f; unsigned u; } c; c.f = f;
    unsigned u = c.u;
    u += 0x7FFFu + ((u >> 16) & 1u);          // round-to-nearest-even
    return (unsigned short)(u >> 16);
}
__device__ __forceinline__ float bf2f(unsigned short h) {
    union { unsigned u; float f; } c; c.u = ((unsigned)h) << 16;
    return c.f;
}

// tanh(x) = sign(x) * (1-e)/(1+e), e = 2^(-2*log2(e)*|x|). Branch-free:
// v_exp_f32 + v_rcp_f32; exact limits (e->0 => 1).
__device__ __forceinline__ float fast_tanh(float x) {
    float ax = __builtin_fabsf(x);
    float e  = __builtin_amdgcn_exp2f(ax * -2.88539008177793f);   // exp(-2|x|)
    float r  = (1.0f - e) * __builtin_amdgcn_rcpf(1.0f + e);
    return __builtin_copysignf(r, x);
}

// One-time: convert weights to bf16, fuse biases, zero h0.
__global__ void prep_kernel(const float* __restrict__ Wih, const float* __restrict__ Whh,
                            const float* __restrict__ bih, const float* __restrict__ bhh,
                            unsigned short* __restrict__ WihB, unsigned short* __restrict__ WhhB,
                            float* __restrict__ bc, unsigned short* __restrict__ h0) {
    const int NW1 = H_ * I_;
    const int NW2 = H_ * H_;
    const int NB  = H_;
    const int NH  = B_ * H_;
    const int total = NW1 + NW2 + NB + NH;
    for (int i = blockIdx.x * blockDim.x + threadIdx.x; i < total; i += gridDim.x * blockDim.x) {
        int j = i;
        if (j < NW1) { WihB[j] = f2bf(Wih[j]); continue; }
        j -= NW1;
        if (j < NW2) { WhhB[j] = f2bf(Whh[j]); continue; }
        j -= NW2;
        if (j < NB)  { bc[j] = bih[j] + bhh[j]; continue; }
        j -= NB;
        h0[j] = 0;
    }
}

// One RNN cell step: h_next = tanh(X @ Wih^T + h @ Whh^T + (b_ih+b_hh))
// mode 0: X[b,k] = xptr[b*xstride + k]                        (warm-up)
// mode 1: X[b,0] = lastptr[b*lastStride]; X[b,k]=ctrl[b,k-1]  (prediction)
__global__ __launch_bounds__(256)
void rnn_step_kernel(int mode,
                     const float* __restrict__ xptr, long xstride,
                     const float* __restrict__ lastptr, long lastStride,
                     const float* __restrict__ ctrlptr,
                     const unsigned short* __restrict__ hcur,
                     unsigned short* __restrict__ hnext,
                     const unsigned short* __restrict__ WihB,
                     const unsigned short* __restrict__ WhhB,
                     const float* __restrict__ bc) {
    // ping-pong staged tiles: [buf][row][32 K + 8 pad] (16B-aligned rows)
    __shared__ __align__(16) unsigned short Ald[2][64][40];
    __shared__ __align__(16) unsigned short Bld[2][64][40];

    const int tid  = threadIdx.x;
    const int lane = tid & 31;
    const int wv   = tid >> 5;
    const int b0   = blockIdx.y * 64;
    const int n0   = blockIdx.x * 64;
    const int mi   = wv & 3;   // wave's 16-row stripe
    const int nj   = wv >> 2;  // wave's pair of 16-col stripes

    // each thread stages one 16B segment per operand per chunk
    const int sr = tid >> 2;          // staging row   0..63
    const int sq = (tid & 3) * 8;     // staging K off 0,8,16,24

    floatx8 acc0 = {};
    floatx8 acc1 = {};

    // ---- staging of one K-chunk into buffer sel ----
    auto stage = [&](int kc) {
        const int kb  = kc * 32;
        const int sel = kc & 1;
        // B: weight rows, B[k][n] = W[n][k]  (always bf16 source)
        if (kb < I_)
            cp16_bf(&WihB[(n0 + sr) * I_ + kb + sq],        &Bld[sel][sr][sq]);
        else
            cp16_bf(&WhhB[(n0 + sr) * H_ + (kb - I_) + sq], &Bld[sel][sr][sq]);
        // A: activations
        if (kb >= I_) {
            cp16_bf(&hcur[(b0 + sr) * H_ + (kb - I_) + sq], &Ald[sel][sr][sq]);
        } else if (mode == 0) {
            const float* src = &xptr[(long)(b0 + sr) * xstride + kb + sq];
            float4 f0 = *(const float4*)src;
            float4 f1 = *(const float4*)(src + 4);
            ushortx8 v;
            v[0] = f2bf(f0.x); v[1] = f2bf(f0.y); v[2] = f2bf(f0.z); v[3] = f2bf(f0.w);
            v[4] = f2bf(f1.x); v[5] = f2bf(f1.y); v[6] = f2bf(f1.z); v[7] = f2bf(f1.w);
            *(ushortx8*)&Ald[sel][sr][sq] = v;
        } else {
            // prediction input: concat([last, controls]) (unaligned, scalar)
            const long b = b0 + sr;
#pragma unroll
            for (int e = 0; e < 8; ++e) {
                const int k = kb + sq + e;
                float f = (k == 0) ? lastptr[b * lastStride]
                                   : ctrlptr[b * (P_ * (I_ - 1)) + (k - 1)];
                Ald[sel][sr][sq + e] = f2bf(f);
            }
        }
    };

    stage(0);
    wait_async_lds();
    __syncthreads();

    for (int kc = 0; kc < NKC; ++kc) {
        if (kc + 1 < NKC) stage(kc + 1);   // overlap with compute of chunk kc
        const int sel = kc & 1;

        // A fragment, 16x32 bf16, this code's own K order (not the ISA table order):
        // lanes 0-15: K 0-15 ; lanes 16-31: K 16-31 (contiguous, matches B)
        FragU af;
        {
            const int m   = mi * 16 + (lane & 15);
            const int klo = (lane < 16) ? 0 : 16;
            af.h[0] = *(const ushortx8*)&Ald[sel][m][klo];
            af.h[1] = *(const ushortx8*)&Ald[sel][m][klo + 8];
        }
        // B fragments: lane holds column N=lane%16, contiguous K 0-15 / 16-31
        FragU bf0, bf1;
        {
            const int kcol = (lane < 16) ? 0 : 16;
            const int nA   = (nj * 2 + 0) * 16 + (lane & 15);
            const int nB   = (nj * 2 + 1) * 16 + (lane & 15);
            bf0.h[0] = *(const ushortx8*)&Bld[sel][nA][kcol];
            bf0.h[1] = *(const ushortx8*)&Bld[sel][nA][kcol + 8];
            bf1.h[0] = *(const ushortx8*)&Bld[sel][nB][kcol];
            bf1.h[1] = *(const ushortx8*)&Bld[sel][nB][kcol + 8];
        }
        acc0 = __builtin_amdgcn_wmma_f32_16x16x32_bf16(false, af.v, false, bf0.v,
                                                       (short)0, acc0, false, false);
        acc1 = __builtin_amdgcn_wmma_f32_16x16x32_bf16(false, af.v, false, bf1.v,
                                                       (short)0, acc1, false, false);
        wait_async_lds();
        __syncthreads();
    }

    // ---- epilogue: bias + fast tanh, store bf16 h_next ----
    // C/D layout: VGPR r, lanes 0-15 -> M=r, N=lane ; lanes 16-31 -> M=r+8, N=lane-16
    const int mrow0 = b0 + mi * 16 + ((lane >> 4) << 3);
    const int ncol  = lane & 15;
    const int nA    = n0 + (nj * 2 + 0) * 16 + ncol;
    const int nB    = n0 + (nj * 2 + 1) * 16 + ncol;
    const float biasA = bc[nA];
    const float biasB = bc[nB];
#pragma unroll
    for (int r = 0; r < 8; ++r) {
        const int brow = mrow0 + r;
        hnext[brow * H_ + nA] = f2bf(fast_tanh(acc0[r] + biasA));
        hnext[brow * H_ + nB] = f2bf(fast_tanh(acc1[r] + biasB));
    }
}

// out[b, t] = h[b,:] . W_fc + b_fc   (one wave32 per batch row)
__global__ __launch_bounds__(256)
void fc_kernel(const unsigned short* __restrict__ h,
               const float* __restrict__ Wfc, const float* __restrict__ bfc,
               float* __restrict__ out, int t) {
    const int lane = threadIdx.x & 31;
    const int b    = blockIdx.x * 8 + (threadIdx.x >> 5);
    float sum = 0.f;
#pragma unroll
    for (int j = 0; j < 32; ++j) {
        const int n = lane + j * 32;
        sum += bf2f(h[b * H_ + n]) * Wfc[n];
    }
    for (int off = 16; off > 0; off >>= 1)
        sum += __shfl_down(sum, off, 32);
    if (lane == 0) out[b * P_ + t] = sum + bfc[0];
}

extern "C" void kernel_launch(void* const* d_in, const int* in_sizes, int n_in,
                              void* d_out, int out_size, void* d_ws, size_t ws_size,
                              hipStream_t stream) {
    const float* past = (const float*)d_in[0];
    const float* fut  = (const float*)d_in[1];
    const float* Wih  = (const float*)d_in[2];
    const float* Whh  = (const float*)d_in[3];
    const float* bih  = (const float*)d_in[4];
    const float* bhh  = (const float*)d_in[5];
    const float* Wfc  = (const float*)d_in[6];
    const float* bfc  = (const float*)d_in[7];
    float* out = (float*)d_out;

    // workspace layout (all 256B aligned), ~3.3 MB total
    char* ws = (char*)d_ws;
    unsigned short* WihB = (unsigned short*)(ws);                                     // 256 KB
    unsigned short* WhhB = (unsigned short*)(ws + 262144);                            // 2 MB
    float*          bc   = (float*)         (ws + 262144 + 2097152);                  // 4 KB
    unsigned short* hA   = (unsigned short*)(ws + 262144 + 2097152 + 4096);           // 512 KB
    unsigned short* hB   = (unsigned short*)(ws + 262144 + 2097152 + 4096 + 524288);  // 512 KB

    prep_kernel<<<2048, 256, 0, stream>>>(Wih, Whh, bih, bhh, WihB, WhhB, bc, hA);

    dim3 grid(H_ / 64, B_ / 64);  // 16 x 4 blocks of 256 threads
    unsigned short* hcur  = hA;
    unsigned short* hnext = hB;

    for (int t = 0; t < S_; ++t) {
        rnn_step_kernel<<<grid, 256, 0, stream>>>(
            0, past + (long)t * I_, (long)S_ * I_,
            nullptr, 0, nullptr,
            hcur, hnext, WihB, WhhB, bc);
        unsigned short* tmp = hcur; hcur = hnext; hnext = tmp;
    }
    for (int t = 0; t < P_; ++t) {
        const float* lastptr    = (t == 0) ? (past + (long)(S_ - 1) * I_) : (out + (t - 1));
        const long   lastStride = (t == 0) ? (long)S_ * I_ : (long)P_;
        rnn_step_kernel<<<grid, 256, 0, stream>>>(
            1, nullptr, 0,
            lastptr, lastStride, fut + (long)t * (I_ - 1),
            hcur, hnext, WihB, WhhB, bc);
        fc_kernel<<<32, 256, 0, stream>>>(hnext, Wfc, bfc, out, t);
        unsigned short* tmp = hcur; hcur = hnext; hnext = tmp;
    }
}
